// GCN_77661598646383
// MI455X (gfx1250) — compile-verified, matrix operand order checked by reading, not run on hardware
//
#include <hip/hip_runtime.h>

#define N_NODES   50000
#define N_EDGES   625000
#define DIM       128
#define N_GRAPHS  128
#define N_CLASSES 10
#define M_TILES   (N_NODES / 16)   // 3125, exact

typedef __attribute__((ext_vector_type(16))) __bf16 v16bf;
typedef __attribute__((ext_vector_type(8)))  float  v8f;

#define LDS_K      256
#define LDS_PAD    8
#define LDS_STRIDE (LDS_K + LDS_PAD)   // 264 bf16 = 528 B/row: 16B-aligned, bank-conflict-free

// ---------------- utility kernels ----------------

__global__ __launch_bounds__(256) void zero4_kernel(float4* __restrict__ p, int n4) {
    int t = blockIdx.x * 256 + threadIdx.x;
    if (t < n4) p[t] = make_float4(0.f, 0.f, 0.f, 0.f);
}

// Build combined transposed bf16 weight: Bt[n][k], k<128 -> W_rel[k][n], else W_root[k-128][n]
__global__ __launch_bounds__(256)
void prep_B_kernel(const float* __restrict__ W_rel, const float* __restrict__ W_root,
                   __bf16* __restrict__ Bt) {
    int t = blockIdx.x * 256 + threadIdx.x;     // 128*256 items
    int n = t >> 8;
    int k = t & 255;
    float v = (k < 128) ? W_rel[k * DIM + n] : W_root[(k - 128) * DIM + n];
    Bt[n * 256 + k] = (__bf16)v;
}

// ---------------- edge gather * w -> scatter-add ----------------

__global__ __launch_bounds__(256)
void edge_scatter_kernel(const float* __restrict__ x, const int* __restrict__ src,
                         const int* __restrict__ dst, const float* __restrict__ ew,
                         float* __restrict__ agg) {
    int t = blockIdx.x * 256 + threadIdx.x;     // N_EDGES*32 items
    int e = t >> 5;                              // wave-uniform
    int q = t & 31;                              // feature quad
    if (e >= N_EDGES) return;
    int   s = src[e];
    int   d = dst[e];
    float w = ew[e];
    float4 v = *(const float4*)(x + (size_t)s * DIM + q * 4);
    float* ap = agg + (size_t)d * DIM + q * 4;
    unsafeAtomicAdd(ap + 0, v.x * w);
    unsafeAtomicAdd(ap + 1, v.y * w);
    unsafeAtomicAdd(ap + 2, v.z * w);
    unsafeAtomicAdd(ap + 3, v.w * w);
}

// ---------------- fused node update: out = relu([agg|x] @ [Wrel;Wroot] + b) ----------------

__global__ __launch_bounds__(256)
void node_update_wmma(const float* __restrict__ agg, const float* __restrict__ xin,
                      const __bf16* __restrict__ Bt,   // [128][256] bf16 (n-major)
                      const float* __restrict__ bias,  // [128]
                      float* __restrict__ out) {
    __shared__ __align__(16) __bf16 lds[16 * LDS_STRIDE];

    const int tid  = threadIdx.x;
    const int lane = tid & 31;
    const int wave = tid >> 5;       // N-tile 0..7
    const int half = lane >> 4;
    const int l15  = lane & 15;

    // B fragments resident in VGPRs (K=256 -> 8 k-steps of 32)
    v16bf bfrag[8];
    {
        const __bf16* brow = Bt + (size_t)(wave * 16 + l15) * 256;
        #pragma unroll
        for (int ks = 0; ks < 8; ++ks) {
            const int k0 = ks * 32 + half * 8;
            union { uint4 u[2]; v16bf v; } cv;
            cv.u[0] = *(const uint4*)(brow + k0);
            cv.u[1] = *(const uint4*)(brow + k0 + 16);
            bfrag[ks] = cv.v;
        }
    }
    const float bn = bias[wave * 16 + l15];

    for (int tile = blockIdx.x; tile < M_TILES; tile += gridDim.x) {
        const int row0 = tile * 16;

        // stage A = [agg | x] (16 rows x 256) into LDS as bf16
        #pragma unroll
        for (int i = 0; i < 4; ++i) {
            int idx = tid + i * 256;            // float4 index, 0..1023
            int r   = idx >> 6;                 // row 0..15
            int c4  = idx & 63;                 // float4 col 0..63
            const float* sp = (c4 < 32) ? (agg + (size_t)(row0 + r) * DIM + c4 * 4)
                                        : (xin + (size_t)(row0 + r) * DIM + (c4 - 32) * 4);
            float4 v = *(const float4*)sp;
            union { __bf16 b[4]; uint2 u; } pk;
            pk.b[0] = (__bf16)v.x; pk.b[1] = (__bf16)v.y;
            pk.b[2] = (__bf16)v.z; pk.b[3] = (__bf16)v.w;
            *(uint2*)&lds[r * LDS_STRIDE + c4 * 4] = pk.u;
        }
        __syncthreads();

        v8f c = {};
        #pragma unroll
        for (int ks = 0; ks < 8; ++ks) {
            const int k0 = ks * 32 + half * 8;
            union { uint4 u[2]; v16bf v; } av;
            av.u[0] = *(const uint4*)&lds[l15 * LDS_STRIDE + k0];
            av.u[1] = *(const uint4*)&lds[l15 * LDS_STRIDE + k0 + 16];
            c = __builtin_amdgcn_wmma_f32_16x16x32_bf16(false, av.v, false, bfrag[ks],
                                                        (short)0, c, false, false);
        }
        __syncthreads();

        // epilogue: C/D layout -> VGPR r is M=r+half*8, lane%16 is N
        #pragma unroll
        for (int r = 0; r < 8; ++r) {
            int   m = r + half * 8;
            float v = c[r] + bn;
            v = v > 0.f ? v : 0.f;
            out[(size_t)(row0 + m) * DIM + wave * 16 + l15] = v;
        }
    }
}

// ---------------- global add pool ----------------

__global__ __launch_bounds__(256)
void pool_kernel(const float* __restrict__ h, const int* __restrict__ batch,
                 float* __restrict__ emb) {
    int t = blockIdx.x * 256 + threadIdx.x;     // N_NODES*32 items
    int node = t >> 5;
    int q    = t & 31;
    if (node >= N_NODES) return;
    int g = batch[node];
    float4 v = *(const float4*)(h + (size_t)node * DIM + q * 4);
    float* ep = emb + (size_t)g * DIM + q * 4;
    unsafeAtomicAdd(ep + 0, v.x);
    unsafeAtomicAdd(ep + 1, v.y);
    unsafeAtomicAdd(ep + 2, v.z);
    unsafeAtomicAdd(ep + 3, v.w);
}

// ---------------- MLP head (tiny, full fp32) ----------------

__global__ __launch_bounds__(128)
void mlp1_kernel(const float* __restrict__ emb, const float* __restrict__ Wl1,
                 const float* __restrict__ bl1, float* __restrict__ z) {
    int g = blockIdx.x;      // graph
    int n = threadIdx.x;     // out feature
    float acc = bl1[n];
    for (int k = 0; k < DIM; ++k) acc += emb[g * DIM + k] * Wl1[k * DIM + n];
    z[g * DIM + n] = acc > 0.f ? acc : 0.f;
}

__global__ __launch_bounds__(256)
void logits_kernel(const float* __restrict__ z, const float* __restrict__ Wl2,
                   const float* __restrict__ bl2, float* __restrict__ logits) {
    int t = blockIdx.x * 256 + threadIdx.x;     // 128*10
    if (t >= N_GRAPHS * N_CLASSES) return;
    int g = t / N_CLASSES, c = t % N_CLASSES;
    float acc = bl2[c];
    for (int k = 0; k < DIM; ++k) acc += z[g * DIM + k] * Wl2[k * N_CLASSES + c];
    logits[g * N_CLASSES + c] = acc;
}

// ---------------- launch ----------------

extern "C" void kernel_launch(void* const* d_in, const int* in_sizes, int n_in,
                              void* d_out, int out_size, void* d_ws, size_t ws_size,
                              hipStream_t stream) {
    const float* x      = (const float*)d_in[0];
    const int*   ei     = (const int*)d_in[1];
    const int*   batch  = (const int*)d_in[2];
    const float* ew     = (const float*)d_in[3];
    const float* W1_rel = (const float*)d_in[4];
    const float* W1_rt  = (const float*)d_in[5];
    const float* b1     = (const float*)d_in[6];
    const float* W2_rel = (const float*)d_in[7];
    const float* W2_rt  = (const float*)d_in[8];
    const float* b2     = (const float*)d_in[9];
    const float* W3_rel = (const float*)d_in[10];
    const float* W3_rt  = (const float*)d_in[11];
    const float* b3     = (const float*)d_in[12];
    const float* Wl1    = (const float*)d_in[13];
    const float* bl1    = (const float*)d_in[14];
    const float* Wl2    = (const float*)d_in[15];
    const float* bl2    = (const float*)d_in[16];

    const int* src = ei;
    const int* dst = ei + N_EDGES;

    // workspace carve-up
    const size_t featBytes = (size_t)N_NODES * DIM * sizeof(float);   // 25.6 MB
    char* ws   = (char*)d_ws;
    float*  agg = (float*)(ws);
    float*  hA  = (float*)(ws + featBytes);
    float*  hB  = (float*)(ws + 2 * featBytes);
    __bf16* Bt1 = (__bf16*)(ws + 3 * featBytes);
    __bf16* Bt2 = Bt1 + 128 * 256;
    __bf16* Bt3 = Bt2 + 128 * 256;
    float*  z   = (float*)(Bt3 + 128 * 256);

    float* out_logits = (float*)d_out;                     // [128][10]
    float* emb        = (float*)d_out + N_GRAPHS * N_CLASSES; // [128][128]

    const int aggN4    = N_NODES * DIM / 4;                // 1.6M float4
    const int aggZeroG = (aggN4 + 255) / 256;
    const int edgeG    = (N_EDGES * 32) / 256;             // 78125
    const int poolG    = (N_NODES * 32) / 256;             // 6250
    const int gemmG    = 1024;                             // persistent M-tile loop

    // weight prep (bf16, transposed, rel/root fused along K)
    prep_B_kernel<<<128, 256, 0, stream>>>(W1_rel, W1_rt, Bt1);
    prep_B_kernel<<<128, 256, 0, stream>>>(W2_rel, W2_rt, Bt2);
    prep_B_kernel<<<128, 256, 0, stream>>>(W3_rel, W3_rt, Bt3);

    // layer 1
    zero4_kernel<<<aggZeroG, 256, 0, stream>>>((float4*)agg, aggN4);
    edge_scatter_kernel<<<edgeG, 256, 0, stream>>>(x, src, dst, ew, agg);
    node_update_wmma<<<gemmG, 256, 0, stream>>>(agg, x, Bt1, b1, hA);

    // layer 2
    zero4_kernel<<<aggZeroG, 256, 0, stream>>>((float4*)agg, aggN4);
    edge_scatter_kernel<<<edgeG, 256, 0, stream>>>(hA, src, dst, ew, agg);
    node_update_wmma<<<gemmG, 256, 0, stream>>>(agg, hA, Bt2, b2, hB);

    // layer 3
    zero4_kernel<<<aggZeroG, 256, 0, stream>>>((float4*)agg, aggN4);
    edge_scatter_kernel<<<edgeG, 256, 0, stream>>>(hB, src, dst, ew, agg);
    node_update_wmma<<<gemmG, 256, 0, stream>>>(agg, hB, Bt3, b3, hA);

    // pool -> emb (lives in d_out; zero whole output first, every call)
    zero4_kernel<<<(out_size / 4 + 255) / 256, 256, 0, stream>>>((float4*)d_out, out_size / 4);
    pool_kernel<<<poolG, 256, 0, stream>>>(hA, batch, emb);

    // head
    mlp1_kernel<<<N_GRAPHS, 128, 0, stream>>>(emb, Wl1, bl1, z);
    logits_kernel<<<(N_GRAPHS * N_CLASSES + 255) / 256, 256, 0, stream>>>(z, Wl2, bl2, out_logits);
}
